// Attentionlayer_74277164417685
// MI455X (gfx1250) — matrix operand order repaired and checked by reading, not compiled, not run on hardware
//
#include <hip/hip_runtime.h>
#include <hip/hip_bf16.h>
#include <stdint.h>

typedef __bf16 bf16;
typedef __attribute__((ext_vector_type(16))) __bf16 v16bf;
typedef __attribute__((ext_vector_type(8)))  float  v8f;
typedef __attribute__((ext_vector_type(4)))  unsigned int u32x4;
typedef __attribute__((ext_vector_type(8)))  int i32x8;
typedef __attribute__((ext_vector_type(4)))  int i32x4;

#define LQ   2048
#define BATB 2
#define DD   1024
#define HH   16
#define HDIM 64
#define MM   (LQ*BATB)   /* 4096 */
#define N3   (3*DD)      /* 3072 */

union FragB16 { uint4 u[2]; v16bf v; };
union U4BF    { uint4 u; bf16 h[8]; };

__device__ __forceinline__ float rmax16(float v) {
  v = fmaxf(v, __shfl_xor(v, 8, 16));
  v = fmaxf(v, __shfl_xor(v, 4, 16));
  v = fmaxf(v, __shfl_xor(v, 2, 16));
  v = fmaxf(v, __shfl_xor(v, 1, 16));
  return v;
}
__device__ __forceinline__ float rsum16(float v) {
  v += __shfl_xor(v, 8, 16);
  v += __shfl_xor(v, 4, 16);
  v += __shfl_xor(v, 2, 16);
  v += __shfl_xor(v, 1, 16);
  return v;
}

// ---------------------------------------------------------------------------
// TDM: DMA a (tileRows x 32) bf16 tile (row stride = rowStrideElems elements)
// from global memory into LDS (packed row-major, 64B per row).
// Descriptor packing per CDNA5 ISA section 8.3/8.4 (D# group 0 / group 1).
// 6-arg builtin form (clang-23 / therock headers).
// ---------------------------------------------------------------------------
__device__ __forceinline__ void tdm_load_tile_2d(unsigned ldsByte, const void* gptr,
                                                 unsigned tileRows,
                                                 unsigned rowStrideElems) {
  const unsigned long long ga = (unsigned long long)(size_t)gptr;
  u32x4 g0;
  g0[0] = 1u;                                       // count=1, user descriptor
  g0[1] = ldsByte;                                  // lds_addr
  g0[2] = (unsigned)(ga & 0xffffffffu);             // global_addr[31:0]
  g0[3] = (unsigned)((ga >> 32) & 0x1ffffffu)       // global_addr[56:32]
        | (2u << 30);                               // type = 2 ("image")
  i32x8 g1;
  g1[0] = (int)(1u << 16);                          // data_size = 1 (2 bytes)
  g1[1] = (int)((rowStrideElems & 0xffffu) << 16);  // tensor_dim0[15:0]
  g1[2] = (int)(((rowStrideElems >> 16) & 0xffffu)  // tensor_dim0[31:16]
        | ((tileRows & 0xffffu) << 16));            // tensor_dim1[15:0]
  g1[3] = (int)(((tileRows >> 16) & 0xffffu)        // tensor_dim1[31:16]
        | (32u << 16));                             // tile_dim0 = 32 elements
  g1[4] = (int)(tileRows & 0xffffu);                // tile_dim1 ; tile_dim2 = 0
  g1[5] = (int)rowStrideElems;                      // tensor_dim0_stride[31:0]
  g1[6] = 0;                                        // stride hi / dim1_stride
  g1[7] = 0;
  const i32x4 z4 = {0, 0, 0, 0};
  const i32x8 z8 = {0, 0, 0, 0, 0, 0, 0, 0};
  __builtin_amdgcn_tensor_load_to_lds(g0, g1, z4, z4, z8, 0);
}

__global__ void cvt_f32_bf16(const float* __restrict__ src, bf16* __restrict__ dst, int n) {
  int i = blockIdx.x * blockDim.x + threadIdx.x;
  int s = gridDim.x * blockDim.x;
  for (; i < n; i += s) dst[i] = (bf16)src[i];
}

// ---------------------------------------------------------------------------
// QKV projection: qkv[m, j] = X[m,:] . W_in[j,:] + b_in[j]
// TDM double-buffered tiles, 8 waves, wave tile 64x32 (4x2 WMMA accums).
// Epilogue scatters into head-major bf16 Q/K/V (Q scaled by 1/sqrt(64)).
// ---------------------------------------------------------------------------
__global__ __launch_bounds__(256)
void qkv_gemm(const bf16* __restrict__ X, const bf16* __restrict__ W,
              const float* __restrict__ bias,
              bf16* __restrict__ qh, bf16* __restrict__ kh, bf16* __restrict__ vh) {
  __shared__ __align__(16) bf16 As[2][128 * 32];
  __shared__ __align__(16) bf16 Bs[2][128 * 32];

  const int tid = threadIdx.x;
  const int ln  = tid & 31;
  const int wv  = tid >> 5;
  const int bm  = blockIdx.y, bn = blockIdx.x;
  const int wm  = (wv >> 2) * 64;
  const int wn  = (wv & 3) * 32;
  const int r   = ln & 15;
  const int hi  = ln >> 4;

  v8f acc[4][2] = {};

  const bf16* Xt = X + (size_t)(bm * 128) * DD;
  const bf16* Wt = W + (size_t)(bn * 128) * DD;
  const unsigned ldsA0 = (unsigned)(size_t)&As[0][0];
  const unsigned ldsA1 = (unsigned)(size_t)&As[1][0];
  const unsigned ldsB0 = (unsigned)(size_t)&Bs[0][0];
  const unsigned ldsB1 = (unsigned)(size_t)&Bs[1][0];

  const int NT = DD / 32;
  if (wv == 0) {                                   // prime the pipeline
    tdm_load_tile_2d(ldsA0, Xt, 128, DD);
    tdm_load_tile_2d(ldsB0, Wt, 128, DD);
  }

  for (int kt = 0; kt < NT; ++kt) {
    const int p = kt & 1;
    if (wv == 0) __builtin_amdgcn_s_wait_tensorcnt(0);
    __syncthreads();                               // tile kt visible; buf p^1 free
    if (wv == 0 && kt + 1 < NT) {
      tdm_load_tile_2d(p ? ldsA0 : ldsA1, Xt + (kt + 1) * 32, 128, DD);
      tdm_load_tile_2d(p ? ldsB0 : ldsB1, Wt + (kt + 1) * 32, 128, DD);
    }

    FragB16 a[4], b[2];
#pragma unroll
    for (int i = 0; i < 4; ++i) {
      const bf16* pa = &As[p][(wm + i * 16 + r) * 32 + 8 * hi];
      a[i].u[0] = *(const uint4*)pa;
      a[i].u[1] = *(const uint4*)(pa + 16);
    }
#pragma unroll
    for (int j = 0; j < 2; ++j) {
      const bf16* pb = &Bs[p][(wn + j * 16 + r) * 32 + 8 * hi];
      b[j].u[0] = *(const uint4*)pb;
      b[j].u[1] = *(const uint4*)(pb + 16);
    }
#pragma unroll
    for (int i = 0; i < 4; ++i)
#pragma unroll
      for (int j = 0; j < 2; ++j)
        acc[i][j] = __builtin_amdgcn_wmma_f32_16x16x32_bf16(
            false, a[i].v, false, b[j].v, (short)0, acc[i][j], false, false);
    __syncthreads();
  }

  // Epilogue: bias, split into q/k/v, head-major scatter, q scale.
#pragma unroll
  for (int j = 0; j < 2; ++j) {
    const int jg   = bn * 128 + wn + j * 16 + r;      // [0, 3072)
    const float bv = bias[jg];
    const int part = jg >> 10;                        // 0=q 1=k 2=v
    const int dj   = jg & (DD - 1);
    const int h    = dj >> 6, e = dj & 63;
    bf16* dst = (part == 0) ? qh : (part == 1) ? kh : vh;
    const float sc = (part == 0) ? 0.125f : 1.0f;     // 1/sqrt(64)
#pragma unroll
    for (int i = 0; i < 4; ++i)
#pragma unroll
      for (int q = 0; q < 8; ++q) {
        const int m = bm * 128 + wm + i * 16 + q + 8 * hi;  // m = l*B + b
        const int l = m >> 1, bb = m & 1;
        const float v = (acc[i][j][q] + bv) * sc;
        dst[((size_t)(bb * HH + h) * LQ + l) * HDIM + e] = (bf16)v;
      }
  }
}

// ---------------------------------------------------------------------------
// Flash attention: one block = one head x one 128-query tile; 8 waves,
// each wave owns 16 query rows across all 2048 keys (64-key tiles).
// ---------------------------------------------------------------------------
__global__ __launch_bounds__(256)
void attn(const bf16* __restrict__ qh, const bf16* __restrict__ kh,
          const bf16* __restrict__ vh, const float* __restrict__ mask,
          bf16* __restrict__ Zb) {
  __shared__ __align__(16) bf16 Qs[128 * 64];
  __shared__ __align__(16) bf16 Ks[64 * 64];
  __shared__ __align__(16) bf16 Vts[64 * 64];      // transposed: [hd][key]
  __shared__ __align__(16) bf16 Pls[8][16 * 64];   // wave-private P staging

  const int tid = threadIdx.x, ln = tid & 31, wv = tid >> 5;
  const int r = ln & 15, hi = ln >> 4;
  const int qb = blockIdx.x;             // query tile 0..15
  const int head = blockIdx.y;           // 0..31  (= b*H + h)
  const int q0 = qb * 128;
  const size_t hbase = (size_t)head * LQ * HDIM;

  { // load Q tile (128x64) once
    const int row = tid >> 1, seg = (tid & 1) * 32;
    const uint4* g = (const uint4*)(qh + hbase + (size_t)(q0 + row) * HDIM + seg);
    uint4* s = (uint4*)&Qs[row * 64 + seg];
    s[0] = g[0]; s[1] = g[1]; s[2] = g[2]; s[3] = g[3];
  }

  float mrow[8], lrow[8];
  v8f zacc[4] = {};
#pragma unroll
  for (int p = 0; p < 8; ++p) { mrow[p] = -1e30f; lrow[p] = 0.f; }

  const int krow = tid >> 2, kseg = (tid & 3) * 16;

  for (int kb = 0; kb < LQ / 64; ++kb) {
    const int kbase = kb * 64;
    { // cooperative K tile load + V tile transposed load
      const uint4* gk = (const uint4*)(kh + hbase + (size_t)(kbase + krow) * HDIM + kseg);
      uint4* sk = (uint4*)&Ks[krow * 64 + kseg];
      sk[0] = gk[0]; sk[1] = gk[1];
      U4BF v0, v1;
      const uint4* gv = (const uint4*)(vh + hbase + (size_t)(kbase + krow) * HDIM + kseg);
      v0.u = gv[0]; v1.u = gv[1];
#pragma unroll
      for (int u = 0; u < 8; ++u) {
        Vts[(kseg + u) * 64 + krow]     = v0.h[u];
        Vts[(kseg + 8 + u) * 64 + krow] = v1.h[u];
      }
      if (kb + 1 < LQ / 64) {
        __builtin_prefetch(kh + hbase + (size_t)(kbase + 64 + krow) * HDIM + kseg, 0, 1);
        __builtin_prefetch(vh + hbase + (size_t)(kbase + 64 + krow) * HDIM + kseg, 0, 1);
      }
    }
    __syncthreads();

    // S = Q . K^T : 16 rows x 64 keys, K-dim 64 = 2 WMMA steps
    v8f sacc[4] = {};
#pragma unroll
    for (int ks = 0; ks < 2; ++ks) {
      const int kk = ks * 32;
      FragB16 a;
      const bf16* pa = &Qs[(wv * 16 + r) * 64 + kk + 8 * hi];
      a.u[0] = *(const uint4*)pa;
      a.u[1] = *(const uint4*)(pa + 16);
#pragma unroll
      for (int nt = 0; nt < 4; ++nt) {
        FragB16 b;
        const bf16* pb = &Ks[(nt * 16 + r) * 64 + kk + 8 * hi];
        b.u[0] = *(const uint4*)pb;
        b.u[1] = *(const uint4*)(pb + 16);
        sacc[nt] = __builtin_amdgcn_wmma_f32_16x16x32_bf16(
            false, a.v, false, b.v, (short)0, sacc[nt], false, false);
      }
    }

    // mask + online softmax (row stats replicated across each 16-lane half)
    float S[4][8];
#pragma unroll
    for (int nt = 0; nt < 4; ++nt) {
      const int keyg = kbase + nt * 16 + r;
#pragma unroll
      for (int p = 0; p < 8; ++p) {
        const int lq = q0 + wv * 16 + p + 8 * hi;
        S[nt][p] = sacc[nt][p] + mask[(size_t)lq * LQ + keyg];
      }
    }
#pragma unroll
    for (int p = 0; p < 8; ++p) {
      float tm = fmaxf(fmaxf(S[0][p], S[1][p]), fmaxf(S[2][p], S[3][p]));
      tm = rmax16(tm);
      const float mnew = fmaxf(mrow[p], tm);
      const float corr = __expf(mrow[p] - mnew);
      float ps = 0.f;
#pragma unroll
      for (int nt = 0; nt < 4; ++nt) {
        const float e = __expf(S[nt][p] - mnew);
        S[nt][p] = e;
        ps += e;
      }
      ps = rsum16(ps);
      lrow[p] = lrow[p] * corr + ps;
      mrow[p] = mnew;
#pragma unroll
      for (int t = 0; t < 4; ++t) zacc[t][p] = zacc[t][p] * corr;
    }

    // re-layout P (C-layout -> A-layout) through wave-private LDS
#pragma unroll
    for (int nt = 0; nt < 4; ++nt)
#pragma unroll
      for (int p = 0; p < 8; ++p)
        Pls[wv][(p + 8 * hi) * 64 + nt * 16 + r] = (bf16)S[nt][p];

    // Z += P . V  (V transposed in LDS so B-fragments are contiguous)
#pragma unroll
    for (int ks = 0; ks < 2; ++ks) {
      const int kk = ks * 32;
      FragB16 a;
      const bf16* pa = &Pls[wv][r * 64 + kk + 8 * hi];
      a.u[0] = *(const uint4*)pa;
      a.u[1] = *(const uint4*)(pa + 16);
#pragma unroll
      for (int t = 0; t < 4; ++t) {
        FragB16 b;
        const bf16* pb = &Vts[(t * 16 + r) * 64 + kk + 8 * hi];
        b.u[0] = *(const uint4*)pb;
        b.u[1] = *(const uint4*)(pb + 16);
        zacc[t] = __builtin_amdgcn_wmma_f32_16x16x32_bf16(
            false, a.v, false, b.v, (short)0, zacc[t], false, false);
      }
    }
    __syncthreads();
  }

  // finalize: Z /= l, store bf16 in (l*B+b, D) layout for output GEMM
  const int bb = head >> 4, h = head & 15;
#pragma unroll
  for (int t = 0; t < 4; ++t)
#pragma unroll
    for (int p = 0; p < 8; ++p) {
      const int l = q0 + wv * 16 + p + 8 * hi;
      const float v = zacc[t][p] / lrow[p];
      Zb[((size_t)(l * BATB + bb)) * DD + h * HDIM + t * 16 + r] = (bf16)v;
    }
}

// ---------------------------------------------------------------------------
// Output projection: out[m, j] = Z[m,:] . W_out[j,:] + b_out[j]   (fp32 out)
// Same TDM double-buffered structure as qkv_gemm.
// ---------------------------------------------------------------------------
__global__ __launch_bounds__(256)
void out_gemm(const bf16* __restrict__ Z, const bf16* __restrict__ W,
              const float* __restrict__ bias, float* __restrict__ out) {
  __shared__ __align__(16) bf16 As[2][128 * 32];
  __shared__ __align__(16) bf16 Bs[2][128 * 32];

  const int tid = threadIdx.x;
  const int ln  = tid & 31;
  const int wv  = tid >> 5;
  const int bm  = blockIdx.y, bn = blockIdx.x;
  const int wm  = (wv >> 2) * 64;
  const int wn  = (wv & 3) * 32;
  const int r   = ln & 15;
  const int hi  = ln >> 4;

  v8f acc[4][2] = {};

  const bf16* Zt = Z + (size_t)(bm * 128) * DD;
  const bf16* Wt = W + (size_t)(bn * 128) * DD;
  const unsigned ldsA0 = (unsigned)(size_t)&As[0][0];
  const unsigned ldsA1 = (unsigned)(size_t)&As[1][0];
  const unsigned ldsB0 = (unsigned)(size_t)&Bs[0][0];
  const unsigned ldsB1 = (unsigned)(size_t)&Bs[1][0];

  const int NT = DD / 32;
  if (wv == 0) {
    tdm_load_tile_2d(ldsA0, Zt, 128, DD);
    tdm_load_tile_2d(ldsB0, Wt, 128, DD);
  }

  for (int kt = 0; kt < NT; ++kt) {
    const int p = kt & 1;
    if (wv == 0) __builtin_amdgcn_s_wait_tensorcnt(0);
    __syncthreads();
    if (wv == 0 && kt + 1 < NT) {
      tdm_load_tile_2d(p ? ldsA0 : ldsA1, Zt + (kt + 1) * 32, 128, DD);
      tdm_load_tile_2d(p ? ldsB0 : ldsB1, Wt + (kt + 1) * 32, 128, DD);
    }

    FragB16 a[4], b[2];
#pragma unroll
    for (int i = 0; i < 4; ++i) {
      const bf16* pa = &As[p][(wm + i * 16 + r) * 32 + 8 * hi];
      a[i].u[0] = *(const uint4*)pa;
      a[i].u[1] = *(const uint4*)(pa + 16);
    }
#pragma unroll
    for (int j = 0; j < 2; ++j) {
      const bf16* pb = &Bs[p][(wn + j * 16 + r) * 32 + 8 * hi];
      b[j].u[0] = *(const uint4*)pb;
      b[j].u[1] = *(const uint4*)(pb + 16);
    }
#pragma unroll
    for (int i = 0; i < 4; ++i)
#pragma unroll
      for (int j = 0; j < 2; ++j)
        acc[i][j] = __builtin_amdgcn_wmma_f32_16x16x32_bf16(
            false, a[i].v, false, b[j].v, (short)0, acc[i][j], false, false);
    __syncthreads();
  }

#pragma unroll
  for (int j = 0; j < 2; ++j) {
    const int jg   = bn * 128 + wn + j * 16 + r;
    const float bv = bias[jg];
#pragma unroll
    for (int i = 0; i < 4; ++i)
#pragma unroll
      for (int q = 0; q < 8; ++q) {
        const int m = bm * 128 + wm + i * 16 + q + 8 * hi;
        out[(size_t)m * DD + jg] = acc[i][j][q] + bv;
      }
  }
}

// ---------------------------------------------------------------------------
extern "C" void kernel_launch(void* const* d_in, const int* in_sizes, int n_in,
                              void* d_out, int out_size, void* d_ws, size_t ws_size,
                              hipStream_t stream) {
  (void)in_sizes; (void)n_in; (void)out_size; (void)ws_size;
  const float* x     = (const float*)d_in[0];   // q == k == v == x
  const float* mask  = (const float*)d_in[3];
  const float* W_in  = (const float*)d_in[4];
  const float* b_in  = (const float*)d_in[5];
  const float* W_out = (const float*)d_in[6];
  const float* b_out = (const float*)d_in[7];

  char* ws = (char*)d_ws;
  bf16* Xb  = (bf16*)(ws + 0);                       // 8 MB
  bf16* Wib = (bf16*)(ws + 8  * 1024 * 1024);        // 6 MB
  bf16* Wob = (bf16*)(ws + 14 * 1024 * 1024);        // 2 MB
  bf16* Qh  = (bf16*)(ws + 16 * 1024 * 1024);        // 8 MB
  bf16* Kh  = (bf16*)(ws + 24 * 1024 * 1024);        // 8 MB
  bf16* Vh  = (bf16*)(ws + 32 * 1024 * 1024);        // 8 MB
  bf16* Zb  = (bf16*)(ws + 40 * 1024 * 1024);        // 8 MB

  cvt_f32_bf16<<<2048, 256, 0, stream>>>(x,     Xb,  MM * DD);
  cvt_f32_bf16<<<2048, 256, 0, stream>>>(W_in,  Wib, N3 * DD);
  cvt_f32_bf16<<<1024, 256, 0, stream>>>(W_out, Wob, DD * DD);

  qkv_gemm<<<dim3(N3 / 128, MM / 128), 256, 0, stream>>>(Xb, Wib, b_in, Qh, Kh, Vh);
  attn    <<<dim3(LQ / 128, BATB * HH), 256, 0, stream>>>(Qh, Kh, Vh, mask, Zb);
  out_gemm<<<dim3(DD / 128, MM / 128), 256, 0, stream>>>(Zb, Wob, b_out, (float*)d_out);
}
